// End2EndModelGraph_27839978013340
// MI455X (gfx1250) — matrix operand order repaired and checked by reading, not compiled, not run on hardware
//
#include <hip/hip_runtime.h>
#include <hip/hip_bf16.h>
#include <math.h>

#define B_ 256
#define L_ 128
#define H_ 512
#define G4H (4 * H_)   // 2048

typedef __attribute__((ext_vector_type(16))) _Float16 v16h;
typedef __attribute__((ext_vector_type(8)))  _Float16 v8h;
typedef __attribute__((ext_vector_type(8)))  float    v8f;

// ---------------------------------------------------------------------------
// WMMA fragment loaders (CDNA5 wave32 layouts, v_wmma_f32_16x16x32_f16)
// A-matrix 16x32 f16: lane<16 -> M=lane, halves[0..7]=K k0..k0+7, halves[8..15]=K k0+16..k0+23
//                     lane>=16 -> M=lane-16, K ranges shifted +8
// ---------------------------------------------------------------------------
__device__ __forceinline__ v16h load_frag_a(const _Float16* p, int ld, int row0, int k0) {
  int lane  = threadIdx.x & 31;
  int r     = lane & 15;
  int khalf = (lane >> 4) << 3;  // 0 or 8
  const _Float16* base = p + (size_t)(row0 + r) * ld + k0 + khalf;
  v8h lo = *(const v8h*)(base);        // K = k0+khalf .. +7
  v8h hi = *(const v8h*)(base + 16);   // K = k0+khalf+16 .. +23
  v16h out;
#pragma unroll
  for (int j = 0; j < 8; ++j) { out[j] = lo[j]; out[j + 8] = hi[j]; }
  return out;
}

// B-matrix 32x16 f16 (stored as W[N,K] row-major; B[k][n] = W[n0+n][k]):
// lane<16 -> N=lane, halves[0..7]=K k0..k0+7, halves[8..15]=K k0+16..k0+23 ; lane>=16 -> N=lane-16, K ranges shifted +8
__device__ __forceinline__ v16h load_frag_b(const _Float16* p, int ld, int n0, int k0) {
  int lane = threadIdx.x & 31;
  const _Float16* base = p + (size_t)(n0 + (lane & 15)) * ld + k0 + ((lane >> 4) << 3);
  v8h lo = *(const v8h*)(base);
  v8h hi = *(const v8h*)(base + 16);
  v16h out;
#pragma unroll
  for (int j = 0; j < 8; ++j) { out[j] = lo[j]; out[j + 8] = hi[j]; }
  return out;
}

__device__ __forceinline__ float sigm(float x) { return 1.0f / (1.0f + __expf(-x)); }

static __device__ const v8f kVZero = {0.f,0.f,0.f,0.f,0.f,0.f,0.f,0.f};

// ---------------------------------------------------------------------------
// CDNA5 async global->LDS copy (ASYNCcnt-tracked, bypasses VGPRs).
// vdst = VGPR holding LDS byte address, vaddr = 64-bit global address, 16B.
// ---------------------------------------------------------------------------
__device__ __forceinline__ void async_copy16(unsigned lds_addr, const void* gptr) {
  asm volatile("global_load_async_to_lds_b128 %0, %1, off"
               :: "v"(lds_addr), "v"(gptr) : "memory");
}
__device__ __forceinline__ void async_wait0() {
  asm volatile("s_wait_asynccnt 0" ::: "memory");
}

// ---------------------------------------------------------------------------
// f32 -> f16 conversion with optional row padding (for odd leading dims)
// ---------------------------------------------------------------------------
__global__ __launch_bounds__(256)
void convert_pad(const float* src, _Float16* dst, int rows, int src_ld, int dst_ld) {
  int i = blockIdx.x * blockDim.x + threadIdx.x;
  int total = rows * dst_ld;
  if (i >= total) return;
  int r = i / dst_ld, k = i - r * dst_ld;
  dst[i] = (k < src_ld) ? (_Float16)src[(size_t)r * src_ld + k] : (_Float16)0.f;
}

__global__ __launch_bounds__(256)
void init_state(float* c, _Float16* h16x2, float* decin) {
  int i = blockIdx.x * blockDim.x + threadIdx.x;
  if (i < B_ * H_)     c[i] = 0.f;
  if (i < 2 * B_ * H_) h16x2[i] = (_Float16)0.f;
  if (i < 2 * B_)      decin[i] = 0.f;
}

// ---------------------------------------------------------------------------
// Encoder LSTM step, fully fused: z = h@Whᵀ + x·Wi + b, then gate nonlinearity.
// Block: 256 thr = 8 waves, one batch M-tile per block (async-staged in LDS),
// each wave owns one hidden N-tile and computes all 4 gate tiles via WMMA.
// Grid: 16 M-tiles * 4 N-groups = 64 blocks.
// ---------------------------------------------------------------------------
__global__ __launch_bounds__(256)
void enc_step(const int* __restrict__ xs,
              const _Float16* __restrict__ h_in, _Float16* __restrict__ h_out,
              float* __restrict__ c, _Float16* __restrict__ enc16,
              const _Float16* __restrict__ Wh16, const float* __restrict__ Wi,
              const float* __restrict__ bias, int t) {
  __shared__ _Float16 sh[16 * H_];
  int mt   = blockIdx.x >> 2;          // 0..15
  int ng   = blockIdx.x & 3;           // 0..3
  int wave = threadIdx.x >> 5;
  int lane = threadIdx.x & 31;
  int m0   = mt * 16;

  // async global->LDS staging of the 16xH activation tile (16B per transfer)
  {
    const v8h* gsrc = (const v8h*)(h_in + (size_t)m0 * H_);
    v8h* ldst = (v8h*)sh;
    for (int i = threadIdx.x; i < (16 * H_) / 8; i += 256)
      async_copy16((unsigned)(size_t)&ldst[i], gsrc + i);
    async_wait0();
  }
  __syncthreads();

  int nt = ng * 8 + wave;              // 0..31 over H/16
  v8f acc[4];
#pragma unroll
  for (int g = 0; g < 4; ++g) acc[g] = kVZero;

  for (int kk = 0; kk < H_; kk += 32) {
    v16h a = load_frag_a(sh, H_, 0, kk);
#pragma unroll
    for (int g = 0; g < 4; ++g) {
      v16h b = load_frag_b(Wh16, H_, g * H_ + nt * 16, kk);
      acc[g] = __builtin_amdgcn_wmma_f32_16x16x32_f16(
          false, a, false, b, (short)0, acc[g], false, false);
    }
  }

  int n = nt * 16 + (lane & 15);
#pragma unroll
  for (int r = 0; r < 8; ++r) {
    int m  = m0 + r + ((lane >> 4) << 3);      // batch row
    float x  = (float)xs[(size_t)m * L_ + t];
    float zi = acc[0][r] + x * Wi[n]            + bias[n];
    float zf = acc[1][r] + x * Wi[H_ + n]       + bias[H_ + n];
    float zg = acc[2][r] + x * Wi[2 * H_ + n]   + bias[2 * H_ + n];
    float zo = acc[3][r] + x * Wi[3 * H_ + n]   + bias[3 * H_ + n];
    size_t ci = (size_t)m * H_ + n;
    float cn = sigm(zf) * c[ci] + sigm(zi) * tanhf(zg);
    float hn = sigm(zo) * tanhf(cn);
    c[ci] = cn;
    h_out[ci] = (_Float16)hn;
    enc16[((size_t)m * L_ + t) * H_ + n] = (_Float16)hn;
  }
}

// ---------------------------------------------------------------------------
// Decoder LSTM step: z = [context,dec_in]@Wiᵀ + h@Whᵀ + b (K = 512+512 WMMA
// + scalar column), fused gates. Also teacher-forces dec_in for next step.
// ---------------------------------------------------------------------------
__global__ __launch_bounds__(256)
void dec_step(const int* __restrict__ xs, const int* __restrict__ argsort,
              const _Float16* __restrict__ ctx16,
              const _Float16* __restrict__ h_in, _Float16* __restrict__ h_out,
              float* __restrict__ c,
              const _Float16* __restrict__ Wi16 /* ld 520 */,
              const _Float16* __restrict__ Wh16,
              const float* __restrict__ Wi_f32 /* [4H,513] */,
              const float* __restrict__ bias,
              const float* __restrict__ dec_in_cur, float* __restrict__ dec_in_next,
              int t) {
  __shared__ _Float16 sc[16 * H_];
  __shared__ _Float16 sh[16 * H_];
  int mt   = blockIdx.x >> 2;
  int ng   = blockIdx.x & 3;
  int wave = threadIdx.x >> 5;
  int lane = threadIdx.x & 31;
  int m0   = mt * 16;

  {
    const v8h* gc = (const v8h*)(ctx16 + (size_t)m0 * H_);
    const v8h* gh = (const v8h*)(h_in + (size_t)m0 * H_);
    v8h* lc = (v8h*)sc; v8h* lh = (v8h*)sh;
    for (int i = threadIdx.x; i < (16 * H_) / 8; i += 256) {
      async_copy16((unsigned)(size_t)&lc[i], gc + i);
      async_copy16((unsigned)(size_t)&lh[i], gh + i);
    }
    async_wait0();
  }
  __syncthreads();

  int nt = ng * 8 + wave;
  v8f acc[4];
#pragma unroll
  for (int g = 0; g < 4; ++g) acc[g] = kVZero;

  // K-part 1: context (512) against dec_Wi columns 0..511 (padded ld=520)
  for (int kk = 0; kk < H_; kk += 32) {
    v16h a = load_frag_a(sc, H_, 0, kk);
#pragma unroll
    for (int g = 0; g < 4; ++g) {
      v16h b = load_frag_b(Wi16, 520, g * H_ + nt * 16, kk);
      acc[g] = __builtin_amdgcn_wmma_f32_16x16x32_f16(
          false, a, false, b, (short)0, acc[g], false, false);
    }
  }
  // K-part 2: previous hidden (512) against dec_Wh
  for (int kk = 0; kk < H_; kk += 32) {
    v16h a = load_frag_a(sh, H_, 0, kk);
#pragma unroll
    for (int g = 0; g < 4; ++g) {
      v16h b = load_frag_b(Wh16, H_, g * H_ + nt * 16, kk);
      acc[g] = __builtin_amdgcn_wmma_f32_16x16x32_f16(
          false, a, false, b, (short)0, acc[g], false, false);
    }
  }

  int n = nt * 16 + (lane & 15);
#pragma unroll
  for (int r = 0; r < 8; ++r) {
    int m  = m0 + r + ((lane >> 4) << 3);
    float din = dec_in_cur[m];  // 513th input column handled in fp32
    float zi = acc[0][r] + din * Wi_f32[(size_t)(n) * 513 + 512]            + bias[n];
    float zf = acc[1][r] + din * Wi_f32[(size_t)(H_ + n) * 513 + 512]       + bias[H_ + n];
    float zg = acc[2][r] + din * Wi_f32[(size_t)(2 * H_ + n) * 513 + 512]   + bias[2 * H_ + n];
    float zo = acc[3][r] + din * Wi_f32[(size_t)(3 * H_ + n) * 513 + 512]   + bias[3 * H_ + n];
    size_t ci = (size_t)m * H_ + n;
    float cn = sigm(zf) * c[ci] + sigm(zi) * tanhf(zg);
    float hn = sigm(zo) * tanhf(cn);
    c[ci] = cn;
    h_out[ci] = (_Float16)hn;
  }

  // teacher-forced pointer input for the NEXT step (ping-pong buffer, no race)
  if (blockIdx.x == 0) {
    int b = threadIdx.x;  // 256 threads == B_
    int idx = argsort[(size_t)b * L_ + t];
    dec_in_next[b] = (float)xs[(size_t)b * L_ + idx];
  }
}

// ---------------------------------------------------------------------------
// Generic WMMA GEMM: C[M,N] (f32) = A16[M,K] @ B16[N,K]^T ; one 16x16 tile/wave
// ---------------------------------------------------------------------------
__global__ __launch_bounds__(256)
void wmma_gemm(const _Float16* __restrict__ A, const _Float16* __restrict__ Bm,
               float* __restrict__ C, int Mtiles, int Ntiles, int K,
               int lda, int ldb, int ldc) {
  int tileId = blockIdx.x * 8 + (threadIdx.x >> 5);
  if (tileId >= Mtiles * Ntiles) return;   // wave-uniform exit
  int mt = tileId / Ntiles;
  int nt = tileId - mt * Ntiles;
  v8f acc = kVZero;
  for (int kk = 0; kk < K; kk += 32) {
    v16h a = load_frag_a(A, lda, mt * 16, kk);
    v16h b = load_frag_b(Bm, ldb, nt * 16, kk);
    acc = __builtin_amdgcn_wmma_f32_16x16x32_f16(
        false, a, false, b, (short)0, acc, false, false);
  }
  int lane = threadIdx.x & 31;
  int n = nt * 16 + (lane & 15);
#pragma unroll
  for (int r = 0; r < 8; ++r) {
    int m = mt * 16 + r + ((lane >> 4) << 3);
    C[(size_t)m * ldc + n] = acc[r];
  }
}

// ---------------------------------------------------------------------------
// Fused attention step: one block per batch row b.
// Phase 1: scores[l] = vt . tanh(encw1[b,l,:] + hw2[b,:])      (8 waves x 16 l)
// Phase 2: softmax + log_softmax(-> out[b,t,:])                 (wave 0)
// Phase 3: context[b,h] = sum_l aj[l] * enc_out[b,l,h]          (all threads)
// hw2/vt per-lane values are register-hoisted (invariant over l).
// ---------------------------------------------------------------------------
__global__ __launch_bounds__(256)
void attn_fused(const float* __restrict__ encw1, const float* __restrict__ hw2,
                const float* __restrict__ vt, const _Float16* __restrict__ enc16,
                _Float16* __restrict__ ctx16, float* __restrict__ out, int t) {
  __shared__ float s_scores[L_];
  __shared__ float s_aj[L_];
  int b    = blockIdx.x;
  int wave = threadIdx.x >> 5;
  int lane = threadIdx.x & 31;

  // hoist hw2[b, lane::32] and vt[lane::32] into registers (16 each)
  float wreg[16], vreg[16];
#pragma unroll
  for (int j = 0; j < 16; ++j) {
    wreg[j] = hw2[(size_t)b * H_ + lane + 32 * j];
    vreg[j] = vt[lane + 32 * j];
  }

  // Phase 1: 8 waves cover 128 rows
  for (int l = wave; l < L_; l += 8) {
    const float* e = encw1 + ((size_t)b * L_ + l) * H_;
    float s = 0.f;
#pragma unroll
    for (int j = 0; j < 16; ++j) s += vreg[j] * tanhf(e[lane + 32 * j] + wreg[j]);
#pragma unroll
    for (int m = 16; m; m >>= 1) s += __shfl_xor(s, m, 32);
    if (lane == 0) s_scores[l] = s;
  }
  __syncthreads();

  // Phase 2: wave 0 does softmax / log_softmax over 128
  if (wave == 0) {
    float v[4]; float mx = -3.0e38f;
#pragma unroll
    for (int i = 0; i < 4; ++i) { v[i] = s_scores[lane * 4 + i]; mx = fmaxf(mx, v[i]); }
#pragma unroll
    for (int m = 16; m; m >>= 1) mx = fmaxf(mx, __shfl_xor(mx, m, 32));
    float sum = 0.f;
#pragma unroll
    for (int i = 0; i < 4; ++i) sum += __expf(v[i] - mx);
#pragma unroll
    for (int m = 16; m; m >>= 1) sum += __shfl_xor(sum, m, 32);
    float ls = __logf(sum);
    float* o = out + ((size_t)b * L_ + t) * L_;
#pragma unroll
    for (int i = 0; i < 4; ++i) {
      int l = lane * 4 + i;
      s_aj[l] = __expf(v[i] - mx) / sum;
      o[l] = v[i] - mx - ls;
    }
  }
  __syncthreads();

  // Phase 3: context reduction, 2 hidden channels per thread
#pragma unroll
  for (int rep = 0; rep < 2; ++rep) {
    int h = threadIdx.x + rep * 256;
    const _Float16* e = enc16 + (size_t)b * L_ * H_ + h;
    float s = 0.f;
    for (int l = 0; l < L_; ++l) s += s_aj[l] * (float)e[(size_t)l * H_];
    ctx16[(size_t)b * H_ + h] = (_Float16)s;
  }
}

// ---------------------------------------------------------------------------
extern "C" void kernel_launch(void* const* d_in, const int* in_sizes, int n_in,
                              void* d_out, int out_size, void* d_ws, size_t ws_size,
                              hipStream_t stream) {
  (void)in_sizes; (void)n_in; (void)out_size; (void)ws_size;
  const int*   xs      = (const int*)  d_in[0];
  const int*   argsort = (const int*)  d_in[2];
  const float* enc_Wi  = (const float*)d_in[3];
  const float* enc_Wh  = (const float*)d_in[4];
  const float* enc_b   = (const float*)d_in[5];
  const float* dec_Wi  = (const float*)d_in[6];
  const float* dec_Wh  = (const float*)d_in[7];
  const float* dec_b   = (const float*)d_in[8];
  const float* w1      = (const float*)d_in[9];
  const float* w2      = (const float*)d_in[10];
  const float* vt      = (const float*)d_in[11];
  float* out = (float*)d_out;

  char* ws = (char*)d_ws;
  size_t off = 0;
  auto alloc = [&](size_t bytes) -> void* {
    void* p = ws + off;
    off = (off + bytes + 255) & ~(size_t)255;
    return p;
  };
  _Float16* enc16   = (_Float16*)alloc((size_t)B_ * L_ * H_ * 2);
  float*    encw1   = (float*)   alloc((size_t)B_ * L_ * H_ * 4);
  _Float16* h16     = (_Float16*)alloc((size_t)2 * B_ * H_ * 2);  // ping-pong
  float*    c       = (float*)   alloc((size_t)B_ * H_ * 4);
  float*    hw2     = (float*)   alloc((size_t)B_ * H_ * 4);
  _Float16* ctx16   = (_Float16*)alloc((size_t)B_ * H_ * 2);
  float*    decin   = (float*)   alloc((size_t)2 * B_ * 4);       // ping-pong
  _Float16* encWh16 = (_Float16*)alloc((size_t)G4H * H_ * 2);
  _Float16* decWh16 = (_Float16*)alloc((size_t)G4H * H_ * 2);
  _Float16* decWi16 = (_Float16*)alloc((size_t)G4H * 520 * 2);    // padded ld
  _Float16* w116    = (_Float16*)alloc((size_t)H_ * H_ * 2);
  _Float16* w216    = (_Float16*)alloc((size_t)H_ * H_ * 2);

  // weight conversions to f16
  convert_pad<<<(G4H * H_ + 255) / 256, 256, 0, stream>>>(enc_Wh, encWh16, G4H, H_, H_);
  convert_pad<<<(G4H * H_ + 255) / 256, 256, 0, stream>>>(dec_Wh, decWh16, G4H, H_, H_);
  convert_pad<<<(G4H * 520 + 255) / 256, 256, 0, stream>>>(dec_Wi, decWi16, G4H, 513, 520);
  convert_pad<<<(H_ * H_ + 255) / 256, 256, 0, stream>>>(w1, w116, H_, H_, H_);
  convert_pad<<<(H_ * H_ + 255) / 256, 256, 0, stream>>>(w2, w216, H_, H_, H_);

  init_state<<<(2 * B_ * H_ + 255) / 256, 256, 0, stream>>>(c, h16, decin);

  _Float16* hbuf[2] = {h16, h16 + (size_t)B_ * H_};
  float*    dbuf[2] = {decin, decin + B_};

  // encoder scan
  for (int t = 0; t < L_; ++t) {
    enc_step<<<64, 256, 0, stream>>>(xs, hbuf[t & 1], hbuf[(t + 1) & 1], c, enc16,
                                     encWh16, enc_Wi, enc_b, t);
  }

  // hoisted enc_w1 = enc_out @ w1^T : [32768,512]x[512,512]
  wmma_gemm<<<(2048 * 32) / 8, 256, 0, stream>>>(enc16, w116, encw1,
                                                 2048, 32, H_, H_, H_, H_);

  // decoder scan (attention uses pre-update h): 3 launches per step
  for (int t = 0; t < L_; ++t) {
    wmma_gemm<<<(16 * 32) / 8, 256, 0, stream>>>(hbuf[t & 1], w216, hw2,
                                                 16, 32, H_, H_, H_, H_);
    attn_fused<<<B_, 256, 0, stream>>>(encw1, hw2, vt, enc16, ctx16, out, t);
    dec_step<<<64, 256, 0, stream>>>(xs, argsort, ctx16, hbuf[t & 1], hbuf[(t + 1) & 1],
                                     c, decWi16, decWh16, dec_Wi, dec_b,
                                     dbuf[t & 1], dbuf[(t + 1) & 1], t);
  }
}
